// Conv2DAttention_25769804338
// MI455X (gfx1250) — matrix operand order repaired and checked by reading, not compiled, not run on hardware
//
#include <hip/hip_runtime.h>
#include <hip/hip_bf16.h>
#include <math.h>

// ---------------------------------------------------------------------------
// Fused Conv2D-QKV + flash attention + projection for MI455X (gfx1250).
// All heavy math goes through v_wmma_f32_16x16x32_f16 (wave32 WMMA).
// ---------------------------------------------------------------------------

typedef _Float16 f16;
typedef __attribute__((ext_vector_type(8)))  _Float16 v8h;
typedef __attribute__((ext_vector_type(16))) _Float16 v16h;
typedef __attribute__((ext_vector_type(8)))  float    v8f;

#define NHEAD 8
#define DHEAD 64
#define DIN   256
#define DOUT  256
#define SDIM  1024   // 32*32
#define BATCH 32

static __device__ __forceinline__ v8f vzero() {
  v8f z;
#pragma unroll
  for (int i = 0; i < 8; ++i) z[i] = 0.f;
  return z;
}

static __device__ __forceinline__ v16h combine16(v8h lo, v8h hi) {
  v16h r;
#pragma unroll
  for (int i = 0; i < 8; ++i) { r[i] = lo[i]; r[i + 8] = hi[i]; }
  return r;
}

// A-matrix fragment (16xK row m per lane%16). Per ISA 7.12.2 16-bit A layout:
// lanes 0-15 hold K = [0..7] and [16..23]; lanes 16-31 hold K = [8..15],[24..31].
static __device__ __forceinline__ v16h ld_a(const f16* rowBase, int laneHi) {
  const f16* p = rowBase + laneHi * 8;
  return combine16(*(const v8h*)p, *(const v8h*)(p + 16));
}

// B-matrix fragment (Kx16, column n per lane%16). Same K order as A:
// lanes 0-15 hold K = [0..7],[16..23]; lanes 16-31 hold K = [8..15],[24..31].
static __device__ __forceinline__ v16h ld_b(const f16* rowBase, int laneHi) {
  const f16* p = rowBase + laneHi * 8;
  return combine16(*(const v8h*)p, *(const v8h*)(p + 16));
}

static __device__ __forceinline__ v8f wmma16(v16h a, v16h b, v8f c) {
  return __builtin_amdgcn_wmma_f32_16x16x32_f16(false, a, false, b,
                                                (short)0, c, false, false);
}

// ---------------------------------------------------------------------------
// Prep kernels
// ---------------------------------------------------------------------------

// rel[s][d] = cumsum(tanh(rh))[s/32][d]/sqrt(32) + cumsum(tanh(rw))[s%32][d]/sqrt(32)
__global__ void rel_kernel(const float* __restrict__ rh,
                           const float* __restrict__ rw,
                           float* __restrict__ rel) {
  int d = threadIdx.x;  // 64 threads
  float relh[32], relw[32];
  const float inv = 0.17677669529663687f;  // 1/sqrt(32)
  float acc = 0.f;
#pragma unroll
  for (int i = 0; i < 32; ++i) { acc += tanhf(rh[i * DHEAD + d]); relh[i] = acc * inv; }
  acc = 0.f;
#pragma unroll
  for (int i = 0; i < 32; ++i) { acc += tanhf(rw[i * DHEAD + d]); relw[i] = acc * inv; }
  for (int s = 0; s < SDIM; ++s)
    rel[s * DHEAD + d] = relh[s >> 5] + relw[s & 31];
}

__global__ void convert_w_kernel(const float* __restrict__ w,
                                 f16* __restrict__ wh, int n) {
  int i = blockIdx.x * 256 + threadIdx.x;
  if (i < n) wh[i] = (f16)w[i];
}

// wp[512][256] f32 -> wpt[256][512] f16  (A-side layout for the projection)
__global__ void transpose_wp_kernel(const float* __restrict__ wp,
                                    f16* __restrict__ wpt) {
  int i = blockIdx.x * 256 + threadIdx.x;  // 131072
  int n = i & 255, k = i >> 8;
  wpt[(size_t)n * 512 + k] = (f16)wp[(size_t)k * 256 + n];
}

// x[b][c][s] f32 -> xt[b][s][c] f16, LDS tile transpose (64x64 tiles)
__global__ void transpose_x_kernel(const float* __restrict__ x,
                                   f16* __restrict__ xt) {
  __shared__ f16 tile[64][65];
  int b = blockIdx.z;
  int c0 = blockIdx.y * 64, s0 = blockIdx.x * 64;
  int tid = threadIdx.x;
  const float* src = x + (size_t)b * DIN * SDIM;
#pragma unroll
  for (int r = 0; r < 16; ++r) {
    int idx = r * 256 + tid;
    int cl = idx >> 6, sl = idx & 63;
    tile[cl][sl] = (f16)src[(size_t)(c0 + cl) * SDIM + s0 + sl];
  }
  __syncthreads();
  f16* dst = xt + (size_t)b * SDIM * DIN;
#pragma unroll
  for (int r = 0; r < 16; ++r) {
    int idx = r * 256 + tid;
    int sl = idx >> 6, cl = idx & 63;
    dst[(size_t)(s0 + sl) * DIN + c0 + cl] = tile[cl][sl];
  }
}

// ---------------------------------------------------------------------------
// QKV GEMM: C[o=1536][s=1024] = wh[1536x256] * xt[b]^T + bias, fused scatter:
//  q: (c+bias+rel)*0.125 -> qb[b][h][s][d]   (1/sqrt(DH) folded into q)
//  k: (c+bias+rel)       -> kb[b][h][s][d]
//  v: (c+bias)           -> vt[b][h][d][s]   (transposed for the 2nd attn GEMM)
// workgroup tile 128x128, wave tile 64(o) x 32(s), 64 WMMAs/wave.
// ---------------------------------------------------------------------------
__global__ __launch_bounds__(256) void qkv_kernel(
    const f16* __restrict__ wh, const f16* __restrict__ xt,
    const float* __restrict__ bias, const float* __restrict__ rel,
    f16* __restrict__ qb, f16* __restrict__ kb, f16* __restrict__ vt) {
  int b = blockIdx.z;
  int lane = threadIdx.x & 31, wave = threadIdx.x >> 5;
  int laneHi = lane >> 4, ln = lane & 15;
  int o0 = blockIdx.y * 128 + (wave >> 2) * 64;
  int s0 = blockIdx.x * 128 + (wave & 3) * 32;
  const f16* xrow = xt + (size_t)b * SDIM * DIN;

  v8f acc[4][2];
#pragma unroll
  for (int mt = 0; mt < 4; ++mt)
#pragma unroll
    for (int nt = 0; nt < 2; ++nt) acc[mt][nt] = vzero();

#pragma unroll 2
  for (int kk = 0; kk < DIN; kk += 32) {
    v16h a[4], bf[2];
#pragma unroll
    for (int mt = 0; mt < 4; ++mt)
      a[mt] = ld_a(wh + (size_t)(o0 + mt * 16 + ln) * DIN + kk, laneHi);
#pragma unroll
    for (int nt = 0; nt < 2; ++nt)
      bf[nt] = ld_b(xrow + (size_t)(s0 + nt * 16 + ln) * DIN + kk, laneHi);
#pragma unroll
    for (int mt = 0; mt < 4; ++mt)
#pragma unroll
      for (int nt = 0; nt < 2; ++nt)
        acc[mt][nt] = wmma16(a[mt], bf[nt], acc[mt][nt]);
  }

#pragma unroll
  for (int mt = 0; mt < 4; ++mt) {
#pragma unroll
    for (int nt = 0; nt < 2; ++nt) {
#pragma unroll
      for (int r = 0; r < 8; ++r) {
        int o = o0 + mt * 16 + r + 8 * laneHi;
        int s = s0 + nt * 16 + ln;
        float v = acc[mt][nt][r] + bias[o];
        int which = o >> 9;        // 0=q 1=k 2=v
        int head  = (o >> 6) & 7;
        int d     = o & 63;
        size_t qkidx = (((size_t)b * NHEAD + head) * SDIM + s) * DHEAD + d;
        if (which == 0) {
          qb[qkidx] = (f16)((v + rel[s * DHEAD + d]) * 0.125f);
        } else if (which == 1) {
          kb[qkidx] = (f16)(v + rel[s * DHEAD + d]);
        } else {
          vt[(((size_t)b * NHEAD + head) * DHEAD + d) * SDIM + s] = (f16)v;
        }
      }
    }
  }
}

// ---------------------------------------------------------------------------
// Flash attention: one wave owns 16 query rows, streams K/V in 32-col chunks.
// sim tile lives in accumulators; probabilities routed through a wave-private
// LDS tile to convert C-layout -> A-layout. 256 WMMAs per wave.
// ---------------------------------------------------------------------------
__global__ __launch_bounds__(256) void attn_kernel(
    const f16* __restrict__ qb, const f16* __restrict__ kb,
    const f16* __restrict__ vt, f16* __restrict__ mha) {
  __shared__ f16 ps[8][16][32];  // per-wave P tile [m][k]
  int b = blockIdx.z, head = blockIdx.y;
  int lane = threadIdx.x & 31, wave = threadIdx.x >> 5;
  int laneHi = lane >> 4, ln = lane & 15;
  int s0 = blockIdx.x * 128 + wave * 16;
  size_t bh = (size_t)b * NHEAD + head;

  const f16* qbase = qb + (bh * SDIM + s0 + ln) * DHEAD;
  v16h qa0 = ld_a(qbase, laneHi);        // d = 0..31
  v16h qa1 = ld_a(qbase + 32, laneHi);   // d = 32..63

  float mrun[8], lrun[8], scale[8];
#pragma unroll
  for (int r = 0; r < 8; ++r) { mrun[r] = -1e30f; lrun[r] = 0.f; }
  v8f oacc[4];
#pragma unroll
  for (int dt = 0; dt < 4; ++dt) oacc[dt] = vzero();

  const f16* krow0 = kb + bh * SDIM * DHEAD;
  const f16* vrow0 = vt + bh * DHEAD * SDIM;

  for (int j = 0; j < SDIM; j += 32) {
    v8f st[2] = {vzero(), vzero()};
#pragma unroll
    for (int nt = 0; nt < 2; ++nt) {
      const f16* kbase = krow0 + (size_t)(j + nt * 16 + ln) * DHEAD;
      st[nt] = wmma16(qa0, ld_b(kbase, laneHi), st[nt]);
      st[nt] = wmma16(qa1, ld_b(kbase + 32, laneHi), st[nt]);
    }
    // online softmax over this 32-column chunk (row = 16 lanes of a half-wave)
#pragma unroll
    for (int r = 0; r < 8; ++r) {
      float t = fmaxf(st[0][r], st[1][r]);
#pragma unroll
      for (int mm = 8; mm >= 1; mm >>= 1) t = fmaxf(t, __shfl_xor(t, mm, 32));
      float mnew = fmaxf(mrun[r], t);
      scale[r] = __expf(mrun[r] - mnew);
      mrun[r] = mnew;
      float p0 = __expf(st[0][r] - mnew);
      float p1 = __expf(st[1][r] - mnew);
      float rs = p0 + p1;
#pragma unroll
      for (int mm = 8; mm >= 1; mm >>= 1) rs += __shfl_xor(rs, mm, 32);
      lrun[r] = lrun[r] * scale[r] + rs;
      int m = r + 8 * laneHi;
      ps[wave][m][ln]      = (f16)p0;
      ps[wave][m][16 + ln] = (f16)p1;
    }
#pragma unroll
    for (int dt = 0; dt < 4; ++dt)
#pragma unroll
      for (int r = 0; r < 8; ++r) oacc[dt][r] *= scale[r];

    v16h pa = ld_a(&ps[wave][ln][0], laneHi);  // DS pipe is in-order per wave
#pragma unroll
    for (int dt = 0; dt < 4; ++dt) {
      const f16* vbase = vrow0 + (size_t)(dt * 16 + ln) * SDIM + j;
      oacc[dt] = wmma16(pa, ld_b(vbase, laneHi), oacc[dt]);
    }
  }

  float inv[8];
#pragma unroll
  for (int r = 0; r < 8; ++r) inv[r] = 1.f / lrun[r];
#pragma unroll
  for (int dt = 0; dt < 4; ++dt) {
#pragma unroll
    for (int r = 0; r < 8; ++r) {
      int sq = s0 + r + 8 * laneHi;
      int d  = dt * 16 + ln;
      mha[((size_t)b * SDIM + sq) * (NHEAD * DHEAD) + head * DHEAD + d] =
          (f16)(oacc[dt][r] * inv[r]);
    }
  }
}

// ---------------------------------------------------------------------------
// Projection computed transposed: out[b][C][s] = wpt[C][512] * mha[b][s][512]^T
// -> C-layout columns are s => coalesced f32 stores. 128 WMMAs/wave.
// ---------------------------------------------------------------------------
__global__ __launch_bounds__(256) void proj_kernel(
    const f16* __restrict__ wpt, const f16* __restrict__ mha,
    float* __restrict__ out) {
  int b = blockIdx.z;
  int lane = threadIdx.x & 31, wave = threadIdx.x >> 5;
  int laneHi = lane >> 4, ln = lane & 15;
  int c0 = blockIdx.y * 128 + (wave >> 2) * 64;
  int s0 = blockIdx.x * 128 + (wave & 3) * 32;
  const f16* mrow = mha + (size_t)b * SDIM * (NHEAD * DHEAD);

  v8f acc[4][2];
#pragma unroll
  for (int mt = 0; mt < 4; ++mt)
#pragma unroll
    for (int nt = 0; nt < 2; ++nt) acc[mt][nt] = vzero();

#pragma unroll 2
  for (int kk = 0; kk < NHEAD * DHEAD; kk += 32) {
    v16h a[4], bf[2];
#pragma unroll
    for (int mt = 0; mt < 4; ++mt)
      a[mt] = ld_a(wpt + (size_t)(c0 + mt * 16 + ln) * 512 + kk, laneHi);
#pragma unroll
    for (int nt = 0; nt < 2; ++nt)
      bf[nt] = ld_b(mrow + (size_t)(s0 + nt * 16 + ln) * 512 + kk, laneHi);
#pragma unroll
    for (int mt = 0; mt < 4; ++mt)
#pragma unroll
      for (int nt = 0; nt < 2; ++nt)
        acc[mt][nt] = wmma16(a[mt], bf[nt], acc[mt][nt]);
  }

#pragma unroll
  for (int mt = 0; mt < 4; ++mt) {
#pragma unroll
    for (int nt = 0; nt < 2; ++nt) {
#pragma unroll
      for (int r = 0; r < 8; ++r) {
        int c = c0 + mt * 16 + r + 8 * laneHi;
        int s = s0 + nt * 16 + ln;
        out[((size_t)b * DOUT + c) * SDIM + s] = acc[mt][nt][r];
      }
    }
  }
}

// ---------------------------------------------------------------------------
// Launch
// ---------------------------------------------------------------------------
extern "C" void kernel_launch(void* const* d_in, const int* in_sizes, int n_in,
                              void* d_out, int out_size, void* d_ws, size_t ws_size,
                              hipStream_t stream) {
  (void)in_sizes; (void)n_in; (void)out_size; (void)ws_size;
  const float* x     = (const float*)d_in[0];
  const float* rh    = (const float*)d_in[1];
  const float* rw    = (const float*)d_in[2];
  const float* w_qkv = (const float*)d_in[3];
  const float* b_qkv = (const float*)d_in[4];
  const float* wp    = (const float*)d_in[5];
  float* out = (float*)d_out;

  char* ws = (char*)d_ws;
  float* rel = (float*)(ws + 0);          //   256 KB
  f16* wh    = (f16*)(ws + 262144);       //   768 KB
  f16* wpt   = (f16*)(ws + 1048576);      //   256 KB
  f16* xt    = (f16*)(ws + 1310720);      //    16 MB
  f16* qb    = (f16*)(ws + 18087936);     //    32 MB
  f16* kb    = (f16*)(ws + 51642368);     //    32 MB
  f16* vt    = (f16*)(ws + 85196800);     //    32 MB
  f16* mha   = (f16*)(ws + 118751232);    //    32 MB  (total ~145.3 MB)

  rel_kernel<<<1, 64, 0, stream>>>(rh, rw, rel);
  convert_w_kernel<<<(1536 * 256) / 256, 256, 0, stream>>>(w_qkv, wh, 1536 * 256);
  transpose_wp_kernel<<<(512 * 256) / 256, 256, 0, stream>>>(wp, wpt);
  transpose_x_kernel<<<dim3(16, 4, 32), 256, 0, stream>>>(x, xt);
  qkv_kernel<<<dim3(8, 12, 32), 256, 0, stream>>>(wh, xt, b_qkv, rel, qb, kb, vt);
  attn_kernel<<<dim3(8, 8, 32), 256, 0, stream>>>(qb, kb, vt, mha);
  proj_kernel<<<dim3(8, 2, 32), 256, 0, stream>>>(wpt, mha, out);
}
